// MambaBlock_79018808311895
// MI455X (gfx1250) — matrix operand order repaired and checked by reading, not compiled, not run on hardware
//
#include <hip/hip_runtime.h>

#define D_MODEL 1024
#define D_STATE 16
#define D_CONV 4
#define D_INNER 2048
#define DT_RANK 64
#define BATCH 2
#define SEQ 1024
#define NROWS (BATCH * SEQ)                 // 2048 GEMM rows
#define XPROJ_OUT (DT_RANK + 2 * D_STATE)   // 96

typedef __attribute__((ext_vector_type(16))) __bf16 bf16x16;
typedef __attribute__((ext_vector_type(8)))  float  f32x8;
typedef __attribute__((ext_vector_type(4)))  unsigned int u32x4;

union Frag {
    bf16x16 h;
    struct { u32x4 lo, hi; } v;
};

// ---------------------------------------------------------------- helpers
__device__ __forceinline__ __bf16 to_bf16(float f) {
    union { unsigned short u; __bf16 b; } cv;
    unsigned int x = __float_as_uint(f);
    cv.u = (unsigned short)((x + 0x7FFFu + ((x >> 16) & 1u)) >> 16);  // RNE
    return cv.b;
}

__device__ __forceinline__ float silu(float v) {
    return v * (1.0f / (1.0f + __expf(-v)));
}

// ---------------------------------------------------------------- f32 -> bf16
__global__ void cvt_bf16_kernel(const float* __restrict__ in,
                                __bf16* __restrict__ out, int n) {
    int i = blockIdx.x * blockDim.x + threadIdx.x;
    if (i < n) out[i] = to_bf16(in[i]);
}

// ---------------------------------------------------------------- LayerNorm
// x: (B, D_MODEL, L) -> xn bf16: (B*L, D_MODEL) row-major (transposed view)
__global__ void layernorm_kernel(const float* __restrict__ x,
                                 const float* __restrict__ w,
                                 const float* __restrict__ b,
                                 __bf16* __restrict__ xn) {
    int row = blockIdx.x;            // b*L + l
    int bb = row >> 10;              // / SEQ
    int l  = row & (SEQ - 1);
    const float* base = x + (size_t)bb * D_MODEL * SEQ + l;  // stride SEQ per ch
    __shared__ float red[256];

    float s = 0.f, s2 = 0.f;
    for (int c = threadIdx.x; c < D_MODEL; c += 256) {
        float v = base[(size_t)c * SEQ];
        s += v; s2 += v * v;
    }
    red[threadIdx.x] = s; __syncthreads();
    for (int st = 128; st > 0; st >>= 1) {
        if (threadIdx.x < st) red[threadIdx.x] += red[threadIdx.x + st];
        __syncthreads();
    }
    float mean = red[0] * (1.0f / D_MODEL);
    __syncthreads();
    red[threadIdx.x] = s2; __syncthreads();
    for (int st = 128; st > 0; st >>= 1) {
        if (threadIdx.x < st) red[threadIdx.x] += red[threadIdx.x + st];
        __syncthreads();
    }
    float var = red[0] * (1.0f / D_MODEL) - mean * mean;
    float rstd = rsqrtf(var + 1e-5f);

    __bf16* orow = xn + (size_t)row * D_MODEL;
    for (int c = threadIdx.x; c < D_MODEL; c += 256) {
        float v = base[(size_t)c * SEQ];
        orow[c] = to_bf16((v - mean) * rstd * w[c] + b[c]);
    }
}

// ---------------------------------------------------------------- WMMA GEMM
// C[M,N] = A[M,K](bf16, LDA) * W[N,K]^T(bf16, LDW), f32 accumulate.
// ALL shapes/strides are compile-time: per-tile load addresses fold into the
// 24-bit instruction IOFFSET, leaving one A pointer + one B pointer live.
// Each wave computes MT M-tiles x NT N-tiles of 16x16 (A reused NT times,
// B reused MT times). Tile coverage is exact — no guards.
// MODE 0: plain f32 store to C
// MODE 1: softplus(acc + bias[n]) store to C
// MODE 2: transposed store with residual: out[(b*NN+n)*LM + l] = acc + resid[.]
template <int MODE, int NWAVES, int NT, int MT,
          int K, int LDA, int LDW, int LDC, int NN, int LM>
__global__ void wmma_gemm_kernel(const __bf16* __restrict__ A,
                                 const __bf16* __restrict__ W,
                                 float* __restrict__ C,
                                 const float* __restrict__ bias,
                                 const float* __restrict__ resid,
                                 float* __restrict__ outT) {
    const int lane = threadIdx.x & 31;
    const int wave = threadIdx.x >> 5;
    const int g    = lane >> 4;               // half-wave group
    const int mr   = lane & 15;
    const int mTile0 = blockIdx.y * MT;       // MT tiles of 16 rows each
    const int nTile0 = (blockIdx.x * NWAVES + wave) * NT;

    f32x8 acc[MT][NT];
    #pragma unroll
    for (int s = 0; s < MT; ++s)
        #pragma unroll
        for (int t = 0; t < NT; ++t)
            acc[s][t] = (f32x8){0.f,0.f,0.f,0.f,0.f,0.f,0.f,0.f};

    // Single A base; M-tile offsets (s*16*LDA) are compile-time immediates.
    // Per lane: halves 0-7 <-> K = k0+8g..+7, halves 8-15 <-> K = k0+16+8g..+7
    const __bf16* aptr = A + (size_t)(mTile0 * 16 + mr) * LDA + 8 * g;
    // Single B base; N-tile offsets (t*16*LDW) are compile-time immediates.
    const __bf16* bptr = W + (size_t)(nTile0 * 16 + mr) * LDW + 8 * g;

    for (int k0 = 0; k0 < K; k0 += 32) {
        Frag afrag[MT];
        #pragma unroll
        for (int s = 0; s < MT; ++s) {
            afrag[s].v.lo = *(const u32x4*)(aptr + s * 16 * LDA + k0);
            afrag[s].v.hi = *(const u32x4*)(aptr + s * 16 * LDA + k0 + 16);
        }
        Frag bfrag[NT];
        #pragma unroll
        for (int t = 0; t < NT; ++t) {
            bfrag[t].v.lo = *(const u32x4*)(bptr + t * 16 * LDW + k0);
            bfrag[t].v.hi = *(const u32x4*)(bptr + t * 16 * LDW + k0 + 16);
        }
        #pragma unroll
        for (int s = 0; s < MT; ++s)
            #pragma unroll
            for (int t = 0; t < NT; ++t)
                acc[s][t] = __builtin_amdgcn_wmma_f32_16x16x32_bf16(
                    false, afrag[s].h, false, bfrag[t].h, (short)0, acc[s][t],
                    false, false);
    }

    #pragma unroll
    for (int s = 0; s < MT; ++s) {
        #pragma unroll
        for (int t = 0; t < NT; ++t) {
            int n = (nTile0 + t) * 16 + mr;
            #pragma unroll
            for (int r = 0; r < 8; ++r) {
                int m = (mTile0 + s) * 16 + r + 8 * g;
                float v = acc[s][t][r];
                if (MODE == 0) {
                    C[(size_t)m * LDC + n] = v;
                } else if (MODE == 1) {
                    v += bias[n];
                    v = (v > 20.f) ? v : log1pf(__expf(v));   // softplus
                    C[(size_t)m * LDC + n] = v;
                } else {  // MODE == 2
                    int b = m / LM, l = m % LM;   // LM power-of-two -> shifts
                    size_t oi = ((size_t)b * NN + n) * LM + l;
                    outT[oi] = v + resid[oi];
                }
            }
        }
    }
}

// ---------------------------------------------------------------- conv + SiLU
// xz: (B*L, 2*D_INNER); x_branch = xz[:, :D_INNER]
// causal depthwise conv over l (kernel 4), + bias, SiLU
__global__ void conv_silu_kernel(const float* __restrict__ xz,
                                 const float* __restrict__ cw,
                                 const float* __restrict__ cb,
                                 float* __restrict__ xc,
                                 __bf16* __restrict__ xcb) {
    int idx = blockIdx.x * blockDim.x + threadIdx.x;  // B*L*D_INNER
    int d  = idx & (D_INNER - 1);
    int bl = idx >> 11;
    int l  = bl & (SEQ - 1);
    int b  = bl >> 10;
    float s = cb[d];
    #pragma unroll
    for (int j = 0; j < D_CONV; ++j) {
        int ls = l - (D_CONV - 1) + j;
        if (ls >= 0)
            s += cw[d * D_CONV + j] *
                 xz[((size_t)(b * SEQ + ls)) * (2 * D_INNER) + d];
    }
    float v = silu(s);
    xc[idx]  = v;
    xcb[idx] = to_bf16(v);
}

// ---------------------------------------------------------------- SSM scan
// Parallel over (b, d); sequential over l. B/C staged in LDS per block.
__global__ void ssm_scan_kernel(const float* __restrict__ xz,     // z branch
                                const float* __restrict__ xconv,  // (B*L,Dinner)
                                const float* __restrict__ xdbl,   // (B*L,96)
                                const float* __restrict__ dtf,    // (B*L,Dinner)
                                const float* __restrict__ A_log,
                                const float* __restrict__ Dvec,
                                __bf16* __restrict__ yout) {
    const int d = blockIdx.x * 256 + threadIdx.x;
    const int b = blockIdx.y;
    __shared__ float sB[D_STATE];
    __shared__ float sC[D_STATE];

    float Ad[D_STATE];
    #pragma unroll
    for (int s = 0; s < D_STATE; ++s)
        Ad[s] = -__expf(A_log[(size_t)d * D_STATE + s]);
    const float Dd = Dvec[d];

    float h[D_STATE];
    #pragma unroll
    for (int s = 0; s < D_STATE; ++s) h[s] = 0.f;

    for (int l = 0; l < SEQ; ++l) {
        const int bl = b * SEQ + l;
        if (threadIdx.x < 2 * D_STATE) {
            int t = threadIdx.x;
            float v = xdbl[(size_t)bl * XPROJ_OUT + DT_RANK + t];
            if (t < D_STATE) sB[t] = v; else sC[t - D_STATE] = v;
        }
        __syncthreads();

        float dt  = dtf[(size_t)bl * D_INNER + d];
        float xv  = xconv[(size_t)bl * D_INNER + d];
        float z   = xz[(size_t)bl * (2 * D_INNER) + D_INNER + d];
        float dtx = dt * xv;
        float y = 0.f;
        #pragma unroll
        for (int s = 0; s < D_STATE; ++s) {
            h[s] = __expf(dt * Ad[s]) * h[s] + dtx * sB[s];
            y += h[s] * sC[s];
        }
        y += xv * Dd;
        y *= silu(z);
        yout[(size_t)bl * D_INNER + d] = to_bf16(y);
        __syncthreads();
    }
}

// ---------------------------------------------------------------- launch
extern "C" void kernel_launch(void* const* d_in, const int* in_sizes, int n_in,
                              void* d_out, int out_size, void* d_ws, size_t ws_size,
                              hipStream_t stream) {
    const float* x         = (const float*)d_in[0];
    const float* ln_w      = (const float*)d_in[1];
    const float* ln_b      = (const float*)d_in[2];
    const float* in_proj_w = (const float*)d_in[3];
    const float* conv_w    = (const float*)d_in[4];
    const float* conv_b    = (const float*)d_in[5];
    const float* x_proj_w  = (const float*)d_in[6];
    const float* dt_proj_w = (const float*)d_in[7];
    const float* dt_proj_b = (const float*)d_in[8];
    const float* A_log     = (const float*)d_in[9];
    const float* Dvec      = (const float*)d_in[10];
    const float* out_proj_w= (const float*)d_in[11];
    float* out = (float*)d_out;

    // ---- workspace layout ----
    char* ws = (char*)d_ws;
    size_t off = 0;
    auto alloc = [&](size_t bytes) {
        char* p = ws + off; off += (bytes + 255) & ~(size_t)255; return p;
    };
    __bf16* wInP  = (__bf16*)alloc((size_t)2 * D_INNER * D_MODEL * 2);   // 4096x1024
    __bf16* wXp   = (__bf16*)alloc((size_t)XPROJ_OUT * D_INNER * 2);     // 96x2048
    __bf16* wDt   = (__bf16*)alloc((size_t)D_INNER * DT_RANK * 2);       // 2048x64
    __bf16* wOut  = (__bf16*)alloc((size_t)D_MODEL * D_INNER * 2);       // 1024x2048
    __bf16* xn    = (__bf16*)alloc((size_t)NROWS * D_MODEL * 2);
    float*  xz    = (float*) alloc((size_t)NROWS * 2 * D_INNER * 4);
    float*  xconv = (float*) alloc((size_t)NROWS * D_INNER * 4);
    __bf16* xconvb= (__bf16*)alloc((size_t)NROWS * D_INNER * 2);
    float*  xdbl  = (float*) alloc((size_t)NROWS * XPROJ_OUT * 4);
    __bf16* xdblb = (__bf16*)alloc((size_t)NROWS * XPROJ_OUT * 2);
    float*  dtf   = (float*) alloc((size_t)NROWS * D_INNER * 4);
    __bf16* yout  = (__bf16*)alloc((size_t)NROWS * D_INNER * 2);
    (void)ws_size; (void)n_in; (void)in_sizes; (void)out_size;

    // ---- weight conversions to bf16 ----
    {
        int n;
        n = 2 * D_INNER * D_MODEL;
        cvt_bf16_kernel<<<(n + 255) / 256, 256, 0, stream>>>(in_proj_w, wInP, n);
        n = XPROJ_OUT * D_INNER;
        cvt_bf16_kernel<<<(n + 255) / 256, 256, 0, stream>>>(x_proj_w, wXp, n);
        n = D_INNER * DT_RANK;
        cvt_bf16_kernel<<<(n + 255) / 256, 256, 0, stream>>>(dt_proj_w, wDt, n);
        n = D_MODEL * D_INNER;
        cvt_bf16_kernel<<<(n + 255) / 256, 256, 0, stream>>>(out_proj_w, wOut, n);
    }

    // ---- LayerNorm ----
    layernorm_kernel<<<NROWS, 256, 0, stream>>>(x, ln_w, ln_b, xn);

    // ---- in_proj: [2048,4096] = xn[2048,1024] * wInP^T ----
    // 4 waves * 4 tiles * 16 = 256 cols/block; 16 blocks = 4096; 2 M-tiles/wave
    wmma_gemm_kernel<0, 4, 4, 2, D_MODEL, D_MODEL, D_MODEL, 2 * D_INNER,
                     2 * D_INNER, 1>
        <<<dim3(16, NROWS / 32), 128, 0, stream>>>(
            xn, wInP, xz, nullptr, nullptr, nullptr);

    // ---- depthwise conv + SiLU ----
    {
        int n = NROWS * D_INNER;
        conv_silu_kernel<<<n / 256, 256, 0, stream>>>(xz, conv_w, conv_b,
                                                      xconv, xconvb);
    }

    // ---- x_proj: [2048,96] = xconv[2048,2048] * wXp^T ----
    // 2 waves * 3 tiles * 16 = 96 cols/block; 1 block covers N exactly
    wmma_gemm_kernel<0, 2, 3, 2, D_INNER, D_INNER, D_INNER, XPROJ_OUT,
                     XPROJ_OUT, 1>
        <<<dim3(1, NROWS / 32), 64, 0, stream>>>(
            xconvb, wXp, xdbl, nullptr, nullptr, nullptr);

    // ---- x_dbl -> bf16 (dt slice consumed as K=64 with lda=96) ----
    {
        int n = NROWS * XPROJ_OUT;
        cvt_bf16_kernel<<<(n + 255) / 256, 256, 0, stream>>>(xdbl, xdblb, n);
    }

    // ---- dt_proj + softplus: [2048,2048] ----
    wmma_gemm_kernel<1, 4, 4, 2, DT_RANK, XPROJ_OUT, DT_RANK, D_INNER,
                     D_INNER, 1>
        <<<dim3(D_INNER / 256, NROWS / 32), 128, 0, stream>>>(
            xdblb, wDt, dtf, dt_proj_b, nullptr, nullptr);

    // ---- SSM scan ----
    ssm_scan_kernel<<<dim3(D_INNER / 256, BATCH), 256, 0, stream>>>(
        xz, xconv, xdbl, dtf, A_log, Dvec, yout);

    // ---- out_proj + transpose + residual ----
    wmma_gemm_kernel<2, 4, 4, 2, D_INNER, D_INNER, D_INNER, 0,
                     D_MODEL, SEQ>
        <<<dim3(D_MODEL / 256, NROWS / 32), 128, 0, stream>>>(
            yout, wOut, nullptr, nullptr, x, out);
}
